// LLoCaAttention_51917564674398
// MI455X (gfx1250) — matrix operand order repaired and checked by reading, not compiled, not run on hardware
//
#include <hip/hip_runtime.h>

typedef _Float16 v16h __attribute__((ext_vector_type(16)));
typedef _Float16 v8h  __attribute__((ext_vector_type(8)));
typedef float    v8f  __attribute__((ext_vector_type(8)));

#define B_    8
#define H_    8
#define N_    1024
#define C_    64
#define HEADS (B_*H_)

// ---------------------------------------------------------------------------
// Kernel 1: per-(b,n) 4x4 inverse + eta-conjugated inverse
// ---------------------------------------------------------------------------
__device__ __forceinline__ float det3x3(const float* m, int r0,int r1,int r2,
                                        int c0,int c1,int c2){
  return m[r0*4+c0]*(m[r1*4+c1]*m[r2*4+c2]-m[r1*4+c2]*m[r2*4+c1])
       - m[r0*4+c1]*(m[r1*4+c0]*m[r2*4+c2]-m[r1*4+c2]*m[r2*4+c0])
       + m[r0*4+c2]*(m[r1*4+c0]*m[r2*4+c1]-m[r1*4+c1]*m[r2*4+c0]);
}

__global__ __launch_bounds__(256)
void frames_inv_kernel(const float* __restrict__ frames,
                       float* __restrict__ inv, float* __restrict__ linv){
  int gid = blockIdx.x*blockDim.x + threadIdx.x;
  if (gid >= B_*N_) return;
  float m[16];
  #pragma unroll
  for (int i=0;i<16;i++) m[i] = frames[(size_t)gid*16 + i];
  float cof[16];
  #pragma unroll
  for (int i=0;i<4;i++){
    int r[3]; { int k=0; for (int t=0;t<4;t++) if (t!=i) r[k++]=t; }
    #pragma unroll
    for (int j=0;j<4;j++){
      int c[3]; { int k=0; for (int t=0;t<4;t++) if (t!=j) c[k++]=t; }
      float d = det3x3(m, r[0],r[1],r[2], c[0],c[1],c[2]);
      cof[i*4+j] = ((i+j)&1) ? -d : d;
    }
  }
  float det = m[0]*cof[0]+m[1]*cof[1]+m[2]*cof[2]+m[3]*cof[3];
  float id  = 1.0f/det;
  #pragma unroll
  for (int i=0;i<4;i++){
    #pragma unroll
    for (int j=0;j<4;j++){
      float e = cof[j*4+i]*id;                 // inv[i][j] = adj/det
      inv [(size_t)gid*16 + i*4+j] = e;
      float sgn = ((i==0)==(j==0)) ? 1.0f : -1.0f;   // eta_i*eta_j
      linv[(size_t)gid*16 + i*4+j] = sgn*e;
    }
  }
}

// ---------------------------------------------------------------------------
// Kernel 2: frame-transform q/k/v, convert to f16.
//  q: inv_frames, scaled by (1/8)*log2(e)   (folded softmax scale, base-2 exp)
//  k: eta*inv*eta
//  v: inv_frames, stored channel-major (transposed) for PV B-fragments
// ---------------------------------------------------------------------------
__global__ __launch_bounds__(256)
void qkv_transform_kernel(const float* __restrict__ q, const float* __restrict__ k,
                          const float* __restrict__ v,
                          const float* __restrict__ inv, const float* __restrict__ linv,
                          _Float16* __restrict__ Qf, _Float16* __restrict__ Kf,
                          _Float16* __restrict__ Vt){
  int gid = blockIdx.x*blockDim.x + threadIdx.x;           // head*N + n
  if (gid >= HEADS*N_) return;
  int head = gid >> 10, n = gid & (N_-1);
  int b = head >> 3;
  size_t mo = ((size_t)b*N_ + n)*16;
  float A[16], L[16];
  #pragma unroll
  for (int i=0;i<16;i++){ A[i]=inv[mo+i]; L[i]=linv[mo+i]; }

  const float QS = 0.125f * 1.44269504088896f;
  const float* qr = q + (size_t)gid*C_;
  const float* kr = k + (size_t)gid*C_;
  const float* vr = v + (size_t)gid*C_;
  _Float16* qo = Qf + (size_t)gid*C_;
  _Float16* ko = Kf + (size_t)gid*C_;

  float vt[C_];
  #pragma unroll
  for (int c=0;c<16;c++){
    qo[c] = (_Float16)(qr[c]*QS);
    ko[c] = (_Float16)(kr[c]);
    vt[c] = vr[c];
  }
  #pragma unroll
  for (int g=0; g<12; g++){
    float q0=qr[16+g*4+0], q1=qr[16+g*4+1], q2=qr[16+g*4+2], q3=qr[16+g*4+3];
    float k0=kr[16+g*4+0], k1=kr[16+g*4+1], k2=kr[16+g*4+2], k3=kr[16+g*4+3];
    float v0=vr[16+g*4+0], v1=vr[16+g*4+1], v2=vr[16+g*4+2], v3=vr[16+g*4+3];
    #pragma unroll
    for (int i=0;i<4;i++){
      float sq = A[i*4+0]*q0 + A[i*4+1]*q1 + A[i*4+2]*q2 + A[i*4+3]*q3;
      float sk = L[i*4+0]*k0 + L[i*4+1]*k1 + L[i*4+2]*k2 + L[i*4+3]*k3;
      float sv = A[i*4+0]*v0 + A[i*4+1]*v1 + A[i*4+2]*v2 + A[i*4+3]*v3;
      qo[16+g*4+i] = (_Float16)(sq*QS);
      ko[16+g*4+i] = (_Float16)(sk);
      vt[16+g*4+i] = sv;
    }
  }
  _Float16* vcol = Vt + (size_t)head*C_*N_;                // [head][ch][n]
  #pragma unroll
  for (int c=0;c<C_;c++)
    vcol[(size_t)c*N_ + n] = (_Float16)vt[c];
}

// ---------------------------------------------------------------------------
// xor-shuffle within each 16-lane half via VALU v_permlane16_b32
// (replaces ds_bpermute: no LDS traffic, co-issues with the matrix pipe)
// sel0 = src nibbles for lanes 0-7, sel1 for lanes 8-15; pattern repeats
// in lanes 16-31. fi=0, bound_ctrl=0.
// ---------------------------------------------------------------------------
__device__ __forceinline__ float plane_xor(float x, unsigned sel0, unsigned sel1){
  unsigned i = (unsigned)__float_as_int(x);
  unsigned r = __builtin_amdgcn_permlane16(i, i, sel0, sel1, false, false);
  return __int_as_float((int)r);
}
#define XOR1(x) plane_xor((x), 0x67452301u, 0xEFCDAB89u)
#define XOR2(x) plane_xor((x), 0x54761032u, 0xDCFE98BAu)
#define XOR4(x) plane_xor((x), 0x32107654u, 0xBA98FEDCu)
#define XOR8(x) plane_xor((x), 0xFEDCBA98u, 0x76543210u)

__device__ __forceinline__ float half_max16(float t){
  t = fmaxf(t, XOR1(t)); t = fmaxf(t, XOR2(t));
  t = fmaxf(t, XOR4(t)); t = fmaxf(t, XOR8(t));
  return t;
}
__device__ __forceinline__ float half_sum16(float t){
  t += XOR1(t); t += XOR2(t); t += XOR4(t); t += XOR8(t);
  return t;
}

// ---------------------------------------------------------------------------
// Kernel 3: flash attention, wave32 WMMA 16x16x32 f16 -> f32
// One wave per 16-row Q tile; 64-key steps (16 WMMAs/step); online softmax.
// ---------------------------------------------------------------------------
#define WMMA_F16(A,B,C) __builtin_amdgcn_wmma_f32_16x16x32_f16( \
        false, (A), false, (B), (short)0, (C), false, false)

__global__ __launch_bounds__(256)
void flash_attn_kernel(const _Float16* __restrict__ Qf, const _Float16* __restrict__ Kf,
                       const _Float16* __restrict__ Vt, float* __restrict__ Out){
  __shared__ __attribute__((aligned(32))) _Float16 plds[8][16*64];
  const int wave = threadIdx.x >> 5;
  const int lane = threadIdx.x & 31;
  const int lh   = lane & 15;
  const int hi   = lane >> 4;
  const int tile = blockIdx.x*8 + wave;        // 4096 tiles
  const int head = tile >> 6;
  const int i0   = (tile & 63) << 4;

  // Q tile as two 16x32 A-fragments (contiguous k order: k = 16*hi + s, same as K)
  v16h qa0, qa1;
  {
    const _Float16* qrow = Qf + ((size_t)head*N_ + i0 + lh)*C_;
    v8h a0 = *(const v8h*)(qrow + hi*16);
    v8h a1 = *(const v8h*)(qrow + hi*16 + 8);
    qa0 = __builtin_shufflevector(a0, a1, 0,1,2,3,4,5,6,7,8,9,10,11,12,13,14,15);
    v8h b0 = *(const v8h*)(qrow + 32 + hi*16);
    v8h b1 = *(const v8h*)(qrow + 32 + hi*16 + 8);
    qa1 = __builtin_shufflevector(b0, b1, 0,1,2,3,4,5,6,7,8,9,10,11,12,13,14,15);
  }

  v8f acc[4] = {{}, {}, {}, {}};
  float mrow[8], lrow[8];
  #pragma unroll
  for (int r=0;r<8;r++){ mrow[r] = -3.0e38f; lrow[r] = 0.0f; }

  const _Float16* kbase = Kf + (size_t)head*N_*C_;
  const _Float16* vbase = Vt + (size_t)head*C_*N_;
  _Float16* pl = plds[wave];

  for (int j0 = 0; j0 < N_; j0 += 64){
    if (j0 + 64 < N_)
      __builtin_prefetch(kbase + (size_t)(j0 + 64 + lh)*C_, 0, 1);

    // ---- S = (Q*scale*log2e) K^T : four 16x16 tiles over 64 keys ----
    v8f s[4];
    #pragma unroll
    for (int jj=0; jj<4; jj++){
      const _Float16* kp = kbase + (size_t)(j0 + jj*16 + lh)*C_ + hi*16;
      v16h kb0 = *(const v16h*)(kp);
      v16h kb1 = *(const v16h*)(kp + 32);
      v8f c = {};
      c = WMMA_F16(qa0, kb0, c);
      c = WMMA_F16(qa1, kb1, c);
      s[jj] = c;
    }

    // ---- online softmax (base-2); one reduction tree per 64 keys ----
    #pragma unroll
    for (int r=0;r<8;r++){
      float t = fmaxf(fmaxf(s[0][r], s[1][r]), fmaxf(s[2][r], s[3][r]));
      t = half_max16(t);
      float mn = fmaxf(mrow[r], t);
      float al = __builtin_amdgcn_exp2f(mrow[r] - mn);
      float e0 = __builtin_amdgcn_exp2f(s[0][r] - mn);
      float e1 = __builtin_amdgcn_exp2f(s[1][r] - mn);
      float e2 = __builtin_amdgcn_exp2f(s[2][r] - mn);
      float e3 = __builtin_amdgcn_exp2f(s[3][r] - mn);
      float rs = half_sum16((e0 + e1) + (e2 + e3));
      lrow[r] = lrow[r]*al + rs;
      mrow[r] = mn;
      acc[0][r]*=al; acc[1][r]*=al; acc[2][r]*=al; acc[3][r]*=al;
      s[0][r]=e0; s[1][r]=e1; s[2][r]=e2; s[3][r]=e3;
    }

    // ---- re-lay P (C-layout) -> two A-fragments via LDS (2KB/wave) ----
    #pragma unroll
    for (int r=0;r<8;r++){
      _Float16* prow = pl + (r + hi*8)*64;
      prow[lh]      = (_Float16)s[0][r];
      prow[16 + lh] = (_Float16)s[1][r];
      prow[32 + lh] = (_Float16)s[2][r];
      prow[48 + lh] = (_Float16)s[3][r];
    }
    v16h pa0, pa1;
    {
      const _Float16* pp = pl + lh*64 + hi*16;
      v8h a0 = *(const v8h*)(pp);
      v8h a1 = *(const v8h*)(pp + 8);
      pa0 = __builtin_shufflevector(a0, a1, 0,1,2,3,4,5,6,7,8,9,10,11,12,13,14,15);
      v8h b0 = *(const v8h*)(pp + 32);
      v8h b1 = *(const v8h*)(pp + 40);
      pa1 = __builtin_shufflevector(b0, b1, 0,1,2,3,4,5,6,7,8,9,10,11,12,13,14,15);
    }

    // ---- O += P V  (V channel-major -> contiguous B fragments) ----
    #pragma unroll
    for (int d4=0; d4<4; d4++){
      const _Float16* vp = vbase + (size_t)(d4*16 + lh)*N_ + j0 + hi*16;
      v16h vb0 = *(const v16h*)(vp);
      v16h vb1 = *(const v16h*)(vp + 32);
      acc[d4] = WMMA_F16(pa0, vb0, acc[d4]);
      acc[d4] = WMMA_F16(pa1, vb1, acc[d4]);
    }
  }

  // ---- normalize + store (C-layout: VGPR r -> row r / r+8) ----
  #pragma unroll
  for (int r=0;r<8;r++){
    float il = 1.0f / lrow[r];
    size_t o = ((size_t)head*N_ + i0 + r + hi*8)*C_ + lh;
    Out[o]      = acc[0][r]*il;
    Out[o + 16] = acc[1][r]*il;
    Out[o + 32] = acc[2][r]*il;
    Out[o + 48] = acc[3][r]*il;
  }
}

// ---------------------------------------------------------------------------
// Kernel 4: apply `frames` to the output, in place (vec4 groups independent)
// ---------------------------------------------------------------------------
__global__ __launch_bounds__(256)
void out_frame_kernel(float* __restrict__ out, const float* __restrict__ frames){
  int gid = blockIdx.x*blockDim.x + threadIdx.x;
  if (gid >= HEADS*N_) return;
  int head = gid >> 10, n = gid & (N_-1), b = head >> 3;
  const float* F = frames + ((size_t)b*N_ + n)*16;
  float M[16];
  #pragma unroll
  for (int i=0;i<16;i++) M[i] = F[i];
  float* o = out + (size_t)gid*C_;
  #pragma unroll
  for (int g=0; g<12; g++){
    float x0=o[16+g*4+0], x1=o[16+g*4+1], x2=o[16+g*4+2], x3=o[16+g*4+3];
    #pragma unroll
    for (int i=0;i<4;i++)
      o[16+g*4+i] = M[i*4+0]*x0 + M[i*4+1]*x1 + M[i*4+2]*x2 + M[i*4+3]*x3;
  }
}

// ---------------------------------------------------------------------------
extern "C" void kernel_launch(void* const* d_in, const int* in_sizes, int n_in,
                              void* d_out, int out_size, void* d_ws, size_t ws_size,
                              hipStream_t stream){
  const float* q  = (const float*)d_in[0];
  const float* k  = (const float*)d_in[1];
  const float* v  = (const float*)d_in[2];
  const float* fr = (const float*)d_in[3];
  float* out = (float*)d_out;

  char* ws = (char*)d_ws;
  float*    inv  = (float*)(ws);                                   // 512 KB
  float*    linv = (float*)(ws + (512u<<10));                      // 512 KB
  _Float16* Qf   = (_Float16*)(ws + (1u<<20));                     // 8 MB
  _Float16* Kf   = (_Float16*)(ws + (1u<<20) + (8u<<20));          // 8 MB
  _Float16* Vt   = (_Float16*)(ws + (1u<<20) + (16u<<20));         // 8 MB

  frames_inv_kernel   <<<(B_*N_)/256,    256, 0, stream>>>(fr, inv, linv);
  qkv_transform_kernel<<<(HEADS*N_)/256, 256, 0, stream>>>(q, k, v, inv, linv, Qf, Kf, Vt);
  flash_attn_kernel   <<<(HEADS*(N_/16))/8, 256, 0, stream>>>(Qf, Kf, Vt, out);
  out_frame_kernel    <<<(HEADS*N_)/256, 256, 0, stream>>>(out, fr);
}
